// ESBN_1683627180490
// MI455X (gfx1250) — matrix operand order repaired and checked by reading, not compiled, not run on hardware
//
#include <hip/hip_runtime.h>

typedef __attribute__((ext_vector_type(16))) __bf16 v16bf;
typedef __attribute__((ext_vector_type(8)))  __bf16 v8bf;
typedef __attribute__((ext_vector_type(8)))  float  v8f;

#define Bsz   4096
#define Tn    64
#define Zn    128
#define Kn    256
#define Hn    512
#define Yn    128
#define KPAD  800     // packed A row: [h(0..511) | key_r(512..768) | zero pad(769..799)]
#define NGATE 2048

static __device__ __forceinline__ float sigm(float x) { return 1.f / (1.f + __expf(-x)); }
static __device__ __forceinline__ __bf16 tobf(float x) { return (__bf16)x; }

// A-fragment (16x32 bf16) per CDNA5 layout:
//   lanes 0-15 : M=lane,    K = {0..7} in chunk0, {16..23} in chunk1
//   lanes 16-31: M=lane-16, K = {8..15} in chunk0, {24..31} in chunk1
static __device__ __forceinline__ v16bf load_a_frag(const __bf16* rowk, int lane) {
  const int ksel = (lane >> 4) << 3;  // 0 or 8
  const v8bf lo = *(const v8bf*)(rowk + ksel);
  const v8bf hi = *(const v8bf*)(rowk + 16 + ksel);
  v16bf a;
#pragma unroll
  for (int i = 0; i < 8; ++i) { a[i] = lo[i]; a[8 + i] = hi[i]; }
  return a;
}

// B fragments pre-packed per-lane-contiguous (32B per lane):
//   flat = (((kt*NT + nt)*32) + lane)*16 + e ;  K = kt*32 + (lane/16)*8 + (e%8) + (e/8)*16 ; N = nt*16 + lane%16

// Read one B fragment from LDS staging buffer: tile f, per-lane 2x16B chunks.
static __device__ __forceinline__ v16bf load_b_sh(const __bf16* sh, int f, int lane) {
  const v8bf lo = *(const v8bf*)(sh + (size_t)((f * 2 + 0) * 32 + lane) * 8);
  const v8bf hi = *(const v8bf*)(sh + (size_t)((f * 2 + 1) * 32 + lane) * 8);
  v16bf b;
#pragma unroll
  for (int i = 0; i < 8; ++i) { b[i] = lo[i]; b[8 + i] = hi[i]; }
  return b;
}

// ---------------- Kernel 1: fused LSTM step GEMM ---------------------------
// gates(128 x 4x64 tile) = A(128x800) @ Wp(800x2048), fused i/f/g/o epilogue.
// B tiles double-buffered in LDS via async global->LDS copies (ASYNCcnt).
// Each wave: 32 rows (2 A frags) x 32 cols x 4 quadrants = 16 WMMA per kt
// against 8 shared B frags -> halves LDS bytes per WMMA vs 16-row waves.
__global__ __launch_bounds__(256) void lstm_step_kernel(
    const __bf16* __restrict__ Ain, __bf16* __restrict__ Aout,
    float* __restrict__ cbuf, const __bf16* __restrict__ Wp,
    const float* __restrict__ b_lstm) {
  const int tid = threadIdx.x;
  const int lane = tid & 31;
  const int wave = tid >> 5;
  const int wm = wave & 3, wn = wave >> 2;
  const int row0 = blockIdx.x * 128 + wm * 32;
  const int colH = blockIdx.y * 64 + wn * 32;  // column within H
  const int nt0 = (blockIdx.y * 64) >> 4;      // first of 4 n-tiles per quadrant
  const int lrow = lane & 15;

  // staging: [buf][tile f=q*4+j][chunk][lane][8 bf16]  (16KB per buffer)
  __shared__ __bf16 Bsh[2][16][2][32][8];

  // Per-thread async-copy state: 4 x 16B chunks per kt, strength-reduced addrs.
  unsigned long long ga[4];
  unsigned ldsoff[4];
  const unsigned shb0 = (unsigned)(uintptr_t)(&Bsh[0][0][0][0][0]);
  const unsigned shb1 = (unsigned)(uintptr_t)(&Bsh[1][0][0][0][0]);
#pragma unroll
  for (int i = 0; i < 4; ++i) {
    const int cid = tid + 256 * i;       // 0..1023
    const int f = cid >> 6;              // tile 0..15
    const int sl = cid & 31;             // lane slot within tile
    const int c = (cid >> 5) & 1;        // 16B chunk within lane's 32B
    const int q = f >> 2, j = f & 3;
    const int ntile = q * (NGATE >> 4) + nt0 + j;
    ga[i] = (unsigned long long)(uintptr_t)(Wp + (size_t)ntile * 512 + sl * 16 + c * 8);
    ldsoff[i] = (unsigned)(((f * 2 + c) * 32 + sl) * 16);  // bytes
  }
  auto stage = [&](int buf) {  // issues kt's tiles, advances ga to next kt
    const unsigned base = buf ? shb1 : shb0;
#pragma unroll
    for (int i = 0; i < 4; ++i) {
      asm volatile("global_load_async_to_lds_b128 %0, %1, off"
                   :: "v"(base + ldsoff[i]), "v"(ga[i]) : "memory");
      ga[i] += (unsigned long long)(NGATE >> 4) * 512 * 2;  // +128 ntiles
    }
  };

  v8f acc[4][2][2] = {};  // [quadrant][ntile][m-subtile]
  const __bf16* arow0 = Ain + (size_t)(row0 + lrow) * KPAD;
  const __bf16* arow1 = Ain + (size_t)(row0 + 16 + lrow) * KPAD;

  stage(0);
  asm volatile("s_wait_asynccnt 0x0" ::: "memory");
  __syncthreads();

  for (int kt = 0; kt < KPAD / 32; ++kt) {
    const int cur = kt & 1;
    if (kt + 1 < KPAD / 32) stage(cur ^ 1);

    const v16bf a0 = load_a_frag(arow0 + kt * 32, lane);
    const v16bf a1 = load_a_frag(arow1 + kt * 32, lane);
    const __bf16* sh = &Bsh[cur][0][0][0][0];
#pragma unroll
    for (int qh = 0; qh < 2; ++qh) {
      v16bf bfr[4];  // batch ds_loads, then WMMAs, so dscnt waits pipeline
#pragma unroll
      for (int u = 0; u < 4; ++u) {
        const int q = qh * 2 + (u >> 1), nt = u & 1;
        bfr[u] = load_b_sh(sh, q * 4 + (wn * 2 + nt), lane);
      }
#pragma unroll
      for (int u = 0; u < 4; ++u) {
        const int q = qh * 2 + (u >> 1), nt = u & 1;
        acc[q][nt][0] = __builtin_amdgcn_wmma_f32_16x16x32_bf16(
            false, a0, false, bfr[u], (short)0, acc[q][nt][0], false, false);
        acc[q][nt][1] = __builtin_amdgcn_wmma_f32_16x16x32_bf16(
            false, a1, false, bfr[u], (short)0, acc[q][nt][1], false, false);
      }
    }

    asm volatile("s_wait_asynccnt 0x0" ::: "memory");
    __syncthreads();
  }

#pragma unroll
  for (int nt = 0; nt < 2; ++nt) {
    const int n = colH + nt * 16 + lrow;  // C/D layout: N = lane%16
    const float bi = b_lstm[n], bf = b_lstm[Hn + n];
    const float bg = b_lstm[2 * Hn + n], bo = b_lstm[3 * Hn + n];
#pragma unroll
    for (int ms = 0; ms < 2; ++ms)
#pragma unroll
      for (int r = 0; r < 8; ++r) {
        const int brow = row0 + ms * 16 + ((lane >> 4) << 3) + r;  // M = r + 8*(lane/16)
        const float ig = sigm(acc[0][nt][ms][r] + bi);
        const float fg = sigm(acc[1][nt][ms][r] + bf);
        const float gg = tanhf(acc[2][nt][ms][r] + bg);
        const float og = sigm(acc[3][nt][ms][r] + bo);
        const size_t cidx = (size_t)brow * Hn + n;
        const float cn = fg * cbuf[cidx] + ig * gg;
        cbuf[cidx] = cn;
        Aout[(size_t)brow * KPAD + n] = tobf(og * tanhf(cn));
      }
  }
}

// ---------------- Kernel 2: key_w = h @ W_kw + b_kw  -> M_k[:,t,:] ---------
__global__ __launch_bounds__(256) void keyw_kernel(
    const __bf16* __restrict__ A, const __bf16* __restrict__ Wp,
    const float* __restrict__ b_kw, __bf16* __restrict__ Mk, int t) {
  const int lane = threadIdx.x & 31;
  const int wave = threadIdx.x >> 5;
  const int wm = wave & 3, wn = wave >> 2;
  const int row0 = blockIdx.x * 64 + wm * 16;
  const int col0 = blockIdx.y * 64 + wn * 32;
  const int lrow = lane & 15;

  v8f acc[2] = {};
  const __bf16* arow = A + (size_t)(row0 + lrow) * KPAD;  // h slice (cols 0..511)
  for (int kt = 0; kt < Hn / 32; ++kt) {
    v16bf a = load_a_frag(arow + kt * 32, lane);
#pragma unroll
    for (int nt = 0; nt < 2; ++nt) {
      const int ntile = (col0 + nt * 16) >> 4;
      v16bf b = *(const v16bf*)(Wp + ((((size_t)kt * (Kn >> 4) + ntile) << 5) + lane) * 16);
      acc[nt] = __builtin_amdgcn_wmma_f32_16x16x32_bf16(
          false, a, false, b, (short)0, acc[nt], false, false);
    }
  }
#pragma unroll
  for (int nt = 0; nt < 2; ++nt) {
    const int n = col0 + nt * 16 + lrow;
    const float bk = b_kw[n];
#pragma unroll
    for (int r = 0; r < 8; ++r) {
      const int brow = row0 + ((lane >> 4) << 3) + r;
      Mk[((size_t)brow * Tn + t) * Kn + n] = tobf(acc[nt][r] + bk);
    }
  }
}

// ---------------- Kernel 3: attention read -> key_r (+ gate g) -------------
__global__ __launch_bounds__(256) void attn_kernel(
    const float* __restrict__ z_seq, const __bf16* __restrict__ Mk,
    __bf16* __restrict__ Abuf, const float* __restrict__ W_g,
    const float* __restrict__ b_g, const float* __restrict__ conf_gain,
    const float* __restrict__ conf_bias, int t) {
  const int b = blockIdx.x;
  const int tid = threadIdx.x;

  if (t == 0) {  // key_r := 0
    Abuf[(size_t)b * KPAD + 512 + tid] = tobf(0.f);
    if (tid == 0) Abuf[(size_t)b * KPAD + 768] = tobf(0.f);
    return;
  }

  __shared__ float zsh[Zn], simsh[Tn], wsh[Tn], cksh[Tn], red[256];
  __shared__ float gsh;

  if (tid < Zn) zsh[tid] = z_seq[((size_t)b * Tn + t) * Zn + tid];

  // g = sigmoid(h . W_g + b_g) ; h = Abuf row b cols 0..511 (bf16)
  float gp = 0.f;
  for (int k = tid; k < Hn; k += 256) gp += (float)Abuf[(size_t)b * KPAD + k] * W_g[k];
  red[tid] = gp;
  __syncthreads();
  for (int s = 128; s > 0; s >>= 1) {
    if (tid < s) red[tid] += red[tid + s];
    __syncthreads();
  }
  if (tid == 0) gsh = sigm(red[0] + b_g[0]);
  __syncthreads();

  // sim[t'] = z_seq[b,t'] . z_t   (M_v[:,t'] == z_seq[:,t'])
  const int lane = tid & 31, w = tid >> 5;
  for (int tp = w; tp < t; tp += 8) {
    const float* zr = z_seq + ((size_t)b * Tn + tp) * Zn;
    float s = 0.f;
    for (int z = lane; z < Zn; z += 32) s += zr[z] * zsh[z];
#pragma unroll
    for (int off = 16; off > 0; off >>= 1) s += __shfl_xor(s, off, 32);
    if (lane == 0) simsh[tp] = s;
  }
  __syncthreads();

  if (tid == 0) {  // softmax over valid t' + confidence
    float m = -1e30f;
    for (int tp = 0; tp < t; ++tp) m = fmaxf(m, simsh[tp]);
    float ssum = 0.f;
    const float cg = conf_gain[0], cb = conf_bias[0];
    for (int tp = 0; tp < t; ++tp) {
      const float e = __expf(simsh[tp] - m);
      wsh[tp] = e;
      ssum += e;
      cksh[tp] = sigm(cg * simsh[tp] + cb);
    }
    red[0] = 1.f / ssum;
  }
  __syncthreads();

  const float inv = red[0], gv = gsh;
  float acc = 0.f;  // read[k] over M_k, one k per thread
  for (int tp = 0; tp < t; ++tp)
    acc += wsh[tp] * (float)Mk[((size_t)b * Tn + tp) * Kn + tid];
  Abuf[(size_t)b * KPAD + 512 + tid] = tobf(gv * inv * acc);
  if (tid == 0) {
    float a2 = 0.f;
    for (int tp = 0; tp < t; ++tp) a2 += wsh[tp] * cksh[tp];
    Abuf[(size_t)b * KPAD + 768] = tobf(gv * inv * a2);
  }
}

// ---------------- Kernel 4: out = h @ W_y + b_y ----------------------------
__global__ __launch_bounds__(256) void out_kernel(
    const __bf16* __restrict__ A, const __bf16* __restrict__ Wp,
    const float* __restrict__ b_y, float* __restrict__ out) {
  const int lane = threadIdx.x & 31;
  const int wave = threadIdx.x >> 5;
  const int wm = wave & 3, wn = wave >> 2;
  const int row0 = blockIdx.x * 64 + wm * 16;
  const int col0 = blockIdx.y * 64 + wn * 32;
  const int lrow = lane & 15;

  v8f acc[2] = {};
  const __bf16* arow = A + (size_t)(row0 + lrow) * KPAD;
  for (int kt = 0; kt < Hn / 32; ++kt) {
    v16bf a = load_a_frag(arow + kt * 32, lane);
#pragma unroll
    for (int nt = 0; nt < 2; ++nt) {
      const int ntile = (col0 + nt * 16) >> 4;
      v16bf b = *(const v16bf*)(Wp + ((((size_t)kt * (Yn >> 4) + ntile) << 5) + lane) * 16);
      acc[nt] = __builtin_amdgcn_wmma_f32_16x16x32_bf16(
          false, a, false, b, (short)0, acc[nt], false, false);
    }
  }
#pragma unroll
  for (int nt = 0; nt < 2; ++nt) {
    const int n = col0 + nt * 16 + lrow;
    const float by = b_y[n];
#pragma unroll
    for (int r = 0; r < 8; ++r) {
      const int brow = row0 + ((lane >> 4) << 3) + r;
      out[(size_t)brow * Yn + n] = acc[nt][r] + by;
    }
  }
}

// ---------------- Setup kernels --------------------------------------------
__global__ void init_kernel(__bf16* A0, __bf16* A1, float* c) {
  const size_t i = (size_t)blockIdx.x * blockDim.x + threadIdx.x;
  if (i < (size_t)Bsz * KPAD) { A0[i] = tobf(0.f); A1[i] = tobf(0.f); }
  if (i < (size_t)Bsz * Hn) c[i] = 0.f;
}

// Pack fp32 weight [rowsA of srcA; rowsB of srcB; zero-pad] (K x N) into the
// per-lane-contiguous B-fragment layout, converted to bf16.
__global__ void pack_b_kernel(__bf16* __restrict__ dst,
                              const float* __restrict__ srcA, int rowsA,
                              const float* __restrict__ srcB, int rowsB,
                              int N, int KT) {
  const size_t idx = (size_t)blockIdx.x * blockDim.x + threadIdx.x;
  const size_t total = (size_t)KT * 32 * N;
  if (idx >= total) return;
  const int e = (int)(idx & 15);
  const int lane = (int)((idx >> 4) & 31);
  const size_t tile = idx >> 9;
  const int nt = (int)(tile % (size_t)(N >> 4));
  const int kt = (int)(tile / (size_t)(N >> 4));
  const int row = kt * 32 + ((lane >> 4) << 3) + (e & 7) + ((e >> 3) << 4);
  const int col = nt * 16 + (lane & 15);
  float v = 0.f;
  if (row < rowsA) v = srcA[(size_t)row * N + col];
  else if (row < rowsA + rowsB) v = srcB[(size_t)(row - rowsA) * N + col];
  dst[idx] = tobf(v);
}

// ---------------- Host side ------------------------------------------------
extern "C" void kernel_launch(void* const* d_in, const int* in_sizes, int n_in,
                              void* d_out, int out_size, void* d_ws, size_t ws_size,
                              hipStream_t stream) {
  const float* z_seq  = (const float*)d_in[0];
  const float* W_i    = (const float*)d_in[1];
  const float* W_h    = (const float*)d_in[2];
  const float* b_lstm = (const float*)d_in[3];
  const float* W_kw   = (const float*)d_in[4];
  const float* b_kw   = (const float*)d_in[5];
  const float* W_g    = (const float*)d_in[6];
  const float* b_g    = (const float*)d_in[7];
  const float* W_y    = (const float*)d_in[8];
  const float* b_y    = (const float*)d_in[9];
  const float* cgain  = (const float*)d_in[10];
  const float* cbias  = (const float*)d_in[11];

  char* p = (char*)d_ws;
  auto take = [&](size_t bytes) -> char* {
    char* r = p;
    p += (bytes + 255) & ~(size_t)255;
    return r;
  };
  __bf16* A0   = (__bf16*)take((size_t)Bsz * KPAD * 2);
  __bf16* A1   = (__bf16*)take((size_t)Bsz * KPAD * 2);
  float*  cbuf = (float*) take((size_t)Bsz * Hn * 4);
  __bf16* Wp   = (__bf16*)take((size_t)KPAD * NGATE * 2);
  __bf16* Wkw  = (__bf16*)take((size_t)Hn * Kn * 2);
  __bf16* Wy   = (__bf16*)take((size_t)Hn * Yn * 2);
  __bf16* Mk   = (__bf16*)take((size_t)Bsz * Tn * Kn * 2);

  {
    const size_t n = (size_t)Bsz * KPAD;
    init_kernel<<<(unsigned)((n + 255) / 256), 256, 0, stream>>>(A0, A1, cbuf);
  }
  pack_b_kernel<<<(unsigned)(((size_t)(KPAD / 32) * 32 * NGATE + 255) / 256), 256, 0, stream>>>(
      Wp, W_h, Hn, W_i, Kn + 1, NGATE, KPAD / 32);
  pack_b_kernel<<<(unsigned)(((size_t)(Hn / 32) * 32 * Kn + 255) / 256), 256, 0, stream>>>(
      Wkw, W_kw, Hn, nullptr, 0, Kn, Hn / 32);
  pack_b_kernel<<<(unsigned)(((size_t)(Hn / 32) * 32 * Yn + 255) / 256), 256, 0, stream>>>(
      Wy, W_y, Hn, nullptr, 0, Yn, Hn / 32);

  for (int t = 0; t < Tn; ++t) {
    const __bf16* Ain = (t & 1) ? A1 : A0;  // step t reads buffer t%2
    __bf16* Aout = (t & 1) ? A0 : A1;       // writes h/key_r into (t+1)%2
    lstm_step_kernel<<<dim3(Bsz / 128, Hn / 64), 256, 0, stream>>>(Ain, Aout, cbuf, Wp, b_lstm);
    keyw_kernel<<<dim3(Bsz / 64, Kn / 64), 256, 0, stream>>>(Aout, Wkw, b_kw, Mk, t);
    attn_kernel<<<Bsz, 256, 0, stream>>>(z_seq, Mk, Aout, W_g, b_g, cgain, cbias, t);
  }
  // final LSTM cell (step 64 reads buffer 0, writes h into buffer 1)
  lstm_step_kernel<<<dim3(Bsz / 128, Hn / 64), 256, 0, stream>>>(A0, A1, cbuf, Wp, b_lstm);
  out_kernel<<<dim3(Bsz / 64, Yn / 64), 256, 0, stream>>>(A1, Wy, b_y, (float*)d_out);
}
